// MoELayer_68264210202900
// MI455X (gfx1250) — matrix operand order repaired and checked by reading, not compiled, not run on hardware
//
#include <hip/hip_runtime.h>
#include <hip/hip_bf16.h>
#include <math.h>

// ---------------- problem constants ----------------
#define HID   1024
#define HID2  2048
#define NEXP  8
#define NTOK  8192            // BATCH*SEQ = 4*2048
#define MT    32              // gathered-token rows per block tile
#define XS_STRIDE (HID + 8)   // LDS row stride (bf16 units), 16B aligned rows
#define HS_STRIDE (HID2 + 8)

typedef __attribute__((ext_vector_type(16))) __bf16 v16bf;
typedef __attribute__((ext_vector_type(8)))  float  v8f;

union FragBF { v16bf v; uint4 q[2]; };

// ---------------- kernel 1: zero output + counts ----------------
__global__ __launch_bounds__(256) void init_kernel(float* __restrict__ out,
                                                   int* __restrict__ cnt) {
    size_t i = ((size_t)blockIdx.x * 256 + threadIdx.x) * 4;   // 8192 blocks cover NTOK*HID exactly
    float4 z = make_float4(0.f, 0.f, 0.f, 0.f);
    *(float4*)(out + i) = z;
    if (blockIdx.x == 0 && threadIdx.x < NEXP) cnt[threadIdx.x] = 0;
}

// ---------------- kernel 2: bf16 transpose-convert of W1, W2 ----------------
// W1t layout: [e][col<2H][k<H]   (so GEMM1 B-fragment K is contiguous)
// W2t layout: [e][col<H][k<2H]
__global__ __launch_bounds__(256) void convert_kernel(const float* __restrict__ W1,
                                                      const float* __restrict__ W2,
                                                      __bf16* __restrict__ W1t,
                                                      __bf16* __restrict__ W2t) {
    size_t i   = (size_t)blockIdx.x * 256 + threadIdx.x;  // < 8*2048*1024 = 2^24
    size_t e   = i >> 21;                                 // H*2H = 2^21 per expert
    size_t rem = i & ((1u << 21) - 1);
    size_t c1 = rem >> 10, r1 = rem & (HID - 1);          // W1t[e][c1][r1]
    W1t[i] = (__bf16)W1[(e * HID + r1) * HID2 + c1];
    size_t c2 = rem >> 11, r2 = rem & (HID2 - 1);         // W2t[e][c2][r2]
    W2t[i] = (__bf16)W2[(e * HID2 + r2) * HID + c2];
}

// ---------------- kernel 3: router (softmax -> top2 -> softmax of top2) ----------------
__global__ __launch_bounds__(256) void router_kernel(const float* __restrict__ x,
                                                     const float* __restrict__ Wr,
                                                     const float* __restrict__ br,
                                                     int*   __restrict__ eTop,
                                                     float* __restrict__ wTop,
                                                     int*   __restrict__ cnt) {
    int wave = threadIdx.x >> 5;
    int lane = threadIdx.x & 31;
    int t = blockIdx.x * 8 + wave;                        // 1024 blocks * 8 waves = 8192 tokens
    const float* xr = x + (size_t)t * HID;
    float acc[NEXP];
    #pragma unroll
    for (int e = 0; e < NEXP; ++e) acc[e] = 0.f;
    for (int h = lane; h < HID; h += 32) {
        float xv = xr[h];
        const float* wr = Wr + (size_t)h * NEXP;
        #pragma unroll
        for (int e = 0; e < NEXP; ++e) acc[e] += xv * wr[e];
    }
    #pragma unroll
    for (int e = 0; e < NEXP; ++e) {
        float v = acc[e];
        for (int m = 16; m >= 1; m >>= 1) v += __shfl_xor(v, m, 32);
        acc[e] = v;
    }
    if (lane == 0) {
        float p[NEXP];
        float mx = -1e30f;
        #pragma unroll
        for (int e = 0; e < NEXP; ++e) { p[e] = acc[e] + br[e]; mx = fmaxf(mx, p[e]); }
        float se = 0.f;
        #pragma unroll
        for (int e = 0; e < NEXP; ++e) { p[e] = expf(p[e] - mx); se += p[e]; }
        float inv = 1.f / se;
        #pragma unroll
        for (int e = 0; e < NEXP; ++e) p[e] *= inv;
        int i1 = 0;
        #pragma unroll
        for (int e = 1; e < NEXP; ++e) if (p[e] > p[i1]) i1 = e;    // lax.top_k tie: lowest index
        int i2 = (i1 == 0) ? 1 : 0;
        #pragma unroll
        for (int e = 0; e < NEXP; ++e) if (e != i1 && p[e] > p[i2]) i2 = e;
        // reference: softmax over the two top *probabilities*
        float tq = expf(p[i2] - p[i1]);
        float w1 = 1.f / (1.f + tq), w2 = tq / (1.f + tq);
        eTop[2 * t]     = i1; wTop[2 * t]     = w1;
        eTop[2 * t + 1] = i2; wTop[2 * t + 1] = w2;
        atomicAdd(&cnt[i1], 1);
        atomicAdd(&cnt[i2], 1);
    }
}

// ---------------- kernel 4: exclusive prefix of counts ----------------
__global__ void prefix_kernel(const int* __restrict__ cnt, int* __restrict__ base,
                              int* __restrict__ cursor) {
    if (threadIdx.x == 0 && blockIdx.x == 0) {
        int s = 0;
        for (int e = 0; e < NEXP; ++e) { base[e] = s; cursor[e] = s; s += cnt[e]; }
    }
}

// ---------------- kernel 5: slot assignment ----------------
__global__ __launch_bounds__(256) void assign_kernel(const int* __restrict__ eTop,
                                                     const float* __restrict__ wTop,
                                                     int* __restrict__ cursor,
                                                     int* __restrict__ tok,
                                                     float* __restrict__ gate) {
    int i = blockIdx.x * 256 + threadIdx.x;               // 64 blocks cover 2*NTOK entries
    int e = eTop[i];
    int idx = atomicAdd(&cursor[e], 1);
    tok[idx]  = i >> 1;
    gate[idx] = wTop[i];
}

// ---------------- kernel 6: fused expert MLP (WMMA bf16) ----------------
__global__ __launch_bounds__(256) void moe_fused(const float* __restrict__ x,
                                                 const float* __restrict__ b1,
                                                 const float* __restrict__ b2,
                                                 const __bf16* __restrict__ W1t,
                                                 const __bf16* __restrict__ W2t,
                                                 const int* __restrict__ cnt,
                                                 const int* __restrict__ base,
                                                 const int* __restrict__ tok,
                                                 const float* __restrict__ gate,
                                                 float* __restrict__ out) {
    const int e    = blockIdx.y;
    const int tile = blockIdx.x;
    const int count = cnt[e];
    if (tile * MT >= count) return;                       // uniform early-exit, before barriers

    extern __shared__ char smem[];
    __bf16* xs = (__bf16*)smem;                                       // MT x XS_STRIDE
    __bf16* hs = (__bf16*)(smem + (size_t)MT * XS_STRIDE * 2);        // MT x HS_STRIDE
    char* meta = smem + (size_t)MT * XS_STRIDE * 2 + (size_t)MT * HS_STRIDE * 2;
    int*   s_tok   = (int*)meta;
    float* s_gate  = (float*)(s_tok + MT);
    int*   s_valid = (int*)(s_gate + MT);

    const int tid = threadIdx.x;
    if (tid < MT) {
        int slot = tile * MT + tid;
        int ok = slot < count;
        int idx = base[e] + (ok ? slot : 0);
        s_tok[tid]   = ok ? tok[idx] : 0;
        s_gate[tid]  = ok ? gate[idx] : 0.f;
        s_valid[tid] = ok;
    }
    __syncthreads();

    // stage gathered x rows -> LDS as bf16 (8 threads per row, 128 floats each)
    {
        int r = tid >> 3, sub = tid & 7;
        const float4* src = (const float4*)(x + (size_t)s_tok[r] * HID) + sub * 32;
        __bf16* dst = xs + r * XS_STRIDE + sub * 128;
        #pragma unroll
        for (int i = 0; i < 32; ++i) {
            float4 f = src[i];
            dst[4 * i + 0] = (__bf16)f.x;
            dst[4 * i + 1] = (__bf16)f.y;
            dst[4 * i + 2] = (__bf16)f.z;
            dst[4 * i + 3] = (__bf16)f.w;
        }
    }
    __syncthreads();

    const int wave = tid >> 5;
    const int lane = tid & 31;
    const int lcol = lane & 15;
    const int hl   = lane >> 4;      // lane half, per ISA 16-bit A/B VGPR maps
    const int mt   = wave & 1;       // 16-row subtile of the 32-row block tile

    // ---- phase 1: h = gelu(x @ W1[e] + b1[e]) into LDS ----
    {
        const int nbase = (wave >> 1) * 512;              // 4 wave-pairs cover 2048 cols
        const __bf16* arow = xs + (mt * 16 + lcol) * XS_STRIDE;
        for (int nt = 0; nt < 32; ++nt) {
            const int ncol = nbase + nt * 16 + lcol;
            const float bias = b1[(size_t)e * HID2 + ncol];
            v8f c = {bias, bias, bias, bias, bias, bias, bias, bias};
            const __bf16* brow = W1t + ((size_t)e * HID2 + ncol) * HID;
            for (int kb = 0; kb < HID; kb += 32) {
                FragBF a, b;
                // A 16x32 bf16: v0-3 => K = 8*hl .. +8 ; v4-7 => K = 16+8*hl .. +8
                a.q[0] = *(const uint4*)(arow + kb + 8 * hl);
                a.q[1] = *(const uint4*)(arow + kb + 16 + 8 * hl);
                // B 32x16 bf16: v0-3 => K = 8*hl .. +8 ; v4-7 => K = 16+8*hl .. +8 (same as A)
                const uint4* bp = (const uint4*)(brow + kb + 8 * hl);
                b.q[0] = bp[0];
                b.q[1] = bp[2];
                __builtin_prefetch(brow + kb + 128, 0, 1);
                c = __builtin_amdgcn_wmma_f32_16x16x32_bf16(false, a.v, false, b.v,
                                                            (short)0, c, false, false);
            }
            #pragma unroll
            for (int v = 0; v < 8; ++v) {                 // C/D: row = v + 8*hl, col = lcol
                float hv = c[v];
                hv = 0.5f * hv * (1.0f + erff(hv * 0.70710678118654752f));  // exact-erf GELU
                hs[(mt * 16 + v + 8 * hl) * HS_STRIDE + ncol] = (__bf16)hv;
            }
        }
    }
    __syncthreads();

    // ---- phase 2: y = h @ W2[e] + b2[e], gate-scale, scatter-add ----
    {
        const int nbase = (wave >> 1) * 256;              // 4 wave-pairs cover 1024 cols
        const __bf16* arow = hs + (mt * 16 + lcol) * HS_STRIDE;
        for (int nt = 0; nt < 16; ++nt) {
            const int ncol = nbase + nt * 16 + lcol;
            const float bias = b2[(size_t)e * HID + ncol];
            v8f c = {bias, bias, bias, bias, bias, bias, bias, bias};
            const __bf16* brow = W2t + ((size_t)e * HID + ncol) * HID2;
            for (int kb = 0; kb < HID2; kb += 32) {
                FragBF a, b;
                a.q[0] = *(const uint4*)(arow + kb + 8 * hl);
                a.q[1] = *(const uint4*)(arow + kb + 16 + 8 * hl);
                const uint4* bp = (const uint4*)(brow + kb + 8 * hl);
                b.q[0] = bp[0];
                b.q[1] = bp[2];
                __builtin_prefetch(brow + kb + 128, 0, 1);
                c = __builtin_amdgcn_wmma_f32_16x16x32_bf16(false, a.v, false, b.v,
                                                            (short)0, c, false, false);
            }
            #pragma unroll
            for (int v = 0; v < 8; ++v) {
                int row = mt * 16 + v + 8 * hl;
                if (s_valid[row]) {                       // exactly 2 commutative adds per out element
                    atomicAdd(out + (size_t)s_tok[row] * HID + ncol, s_gate[row] * c[v]);
                }
            }
        }
    }
}

// ---------------- launch ----------------
extern "C" void kernel_launch(void* const* d_in, const int* in_sizes, int n_in,
                              void* d_out, int out_size, void* d_ws, size_t ws_size,
                              hipStream_t stream) {
    const float* x  = (const float*)d_in[0];
    const float* Wr = (const float*)d_in[1];
    const float* br = (const float*)d_in[2];
    const float* W1 = (const float*)d_in[3];
    const float* b1 = (const float*)d_in[4];
    const float* W2 = (const float*)d_in[5];
    const float* b2 = (const float*)d_in[6];
    float* out = (float*)d_out;

    char* ws = (char*)d_ws;
    const size_t wbytes = (size_t)NEXP * HID2 * HID * 2;  // 33,554,432 B per weight plane
    __bf16* W1t = (__bf16*)ws;
    __bf16* W2t = (__bf16*)(ws + wbytes);
    char* p = ws + 2 * wbytes;
    int*   eTop = (int*)p;    p += (size_t)2 * NTOK * 4;
    float* wTop = (float*)p;  p += (size_t)2 * NTOK * 4;
    int*   tok  = (int*)p;    p += (size_t)2 * NTOK * 4;
    float* gate = (float*)p;  p += (size_t)2 * NTOK * 4;
    int* cnt    = (int*)p;    p += 256;
    int* base   = (int*)p;    p += 256;
    int* cur    = (int*)p;

    init_kernel<<<8192, 256, 0, stream>>>(out, cnt);                       // NTOK*HID floats
    convert_kernel<<<65536, 256, 0, stream>>>(W1, W2, W1t, W2t);           // 2^24 elements
    router_kernel<<<1024, 256, 0, stream>>>(x, Wr, br, eTop, wTop, cnt);
    prefix_kernel<<<1, 32, 0, stream>>>(cnt, base, cur);
    assign_kernel<<<64, 256, 0, stream>>>(eTop, wTop, cur, tok, gate);

    const size_t smem = (size_t)MT * XS_STRIDE * 2 + (size_t)MT * HS_STRIDE * 2 + MT * 12;
    moe_fused<<<dim3(NTOK / MT, NEXP), 256, smem, stream>>>(x, b1, b2, W1t, W2t,
                                                            cnt, base, tok, gate, out);
}
